// CSAModule_47768626266174
// MI455X (gfx1250) — matrix operand order repaired and checked by reading, not compiled, not run on hardware
//
#include <hip/hip_runtime.h>

// Problem constants (match reference)
#define BB 64
#define FF 128
#define TT 512
#define CC 10
#define DD 128

typedef float    v8f    __attribute__((ext_vector_type(8)));
typedef _Float16 half8  __attribute__((ext_vector_type(8)));
typedef _Float16 half16 __attribute__((ext_vector_type(16)));

#define WMMA_F16(a, b, c) \
  __builtin_amdgcn_wmma_f32_16x16x32_f16(false, (a), false, (b), (short)0, (c), false, false)

// ---- workspace layout (bytes, all 256-aligned) ----
#define OFF_INV  0u
#define OFF_WQ   256u
#define OFF_WK   (256u + 32768u)
#define OFF_XCT  (256u + 2u * 32768u)
#define SZ_CTF   ((unsigned)(CC * TT * FF * 2))          // f16 [C][T][F] (D==F)
#define OFF_QH   (OFF_XCT + SZ_CTF)
#define OFF_KH   (OFF_QH + SZ_CTF)
#define OFF_SM   (OFF_KH + SZ_CTF)

// -------- gfx1250 async global->LDS copy (ASYNCcnt-tracked) --------
// ISA 15.18.3: GLOBAL_LOAD_ASYNC_TO_LDS_B128  vdst(=LDS byte offset), vaddr64, off
__device__ __forceinline__ unsigned lds_off_of(const void* p) {
  // generic->local addrspacecast; on gfx1250 flat LDS addr[31:0] is the
  // wave-relative LDS byte address (ISA 10.2), which is what VDST wants.
  return (unsigned)(unsigned long long)(const __attribute__((address_space(3))) void*)p;
}
__device__ __forceinline__ void async_copy_b128(unsigned lds_byte_off, const void* gsrc) {
  asm volatile("global_load_async_to_lds_b128 %0, %1, off"
               :: "v"(lds_byte_off), "v"(gsrc) : "memory");
}
__device__ __forceinline__ void wait_async0() {
  asm volatile("s_wait_asynccnt 0x0" ::: "memory");
}

// -------- WMMA fragment loaders (wave32; contiguous K order, same for A and B) --------
// A-matrix 16x32 f16, M rows, K contiguous (row stride ld halfs).
// lane: m = lane&15, h = lane>>4; element e holds K = h*16 + e
__device__ __forceinline__ half16 frag_a(const _Float16* base, int ld, int lane) {
  int m = lane & 15, h = lane >> 4;
  const _Float16* p = base + (size_t)m * ld + h * 16;
  half8 lo = *(const half8*)p;
  half8 hi = *(const half8*)(p + 8);
  half16 r;
#pragma unroll
  for (int e = 0; e < 8; ++e) { r[e] = lo[e]; r[e + 8] = hi[e]; }
  return r;
}
// B-matrix 32x16 f16, B[k][n]; storage n-major, K contiguous (row stride ld halfs).
// lane: n = lane&15, h = lane>>4; element e holds K = h*16 + e -> one contiguous 32B load.
__device__ __forceinline__ half16 frag_b(const _Float16* base, int ld, int lane) {
  int n = lane & 15, h = lane >> 4;
  return *(const half16*)(base + (size_t)n * ld + h * 16);
}

// -------- kernel 1: class counts + f16 weight conversion --------
__global__ void csa_prep(const float* __restrict__ kw, const float* __restrict__ qw,
                         const int* __restrict__ labels,
                         float* __restrict__ invcnt,
                         _Float16* __restrict__ wk16, _Float16* __restrict__ wq16) {
  int tid = threadIdx.x;
  if (tid < CC) {
    int cnt = 0;
    for (int b = 0; b < BB; ++b) cnt += (labels[b] == tid) ? 1 : 0;
    if (cnt < 1) cnt = 1;
    invcnt[tid] = 1.0f / (float)cnt;
  }
  for (int i = tid; i < DD * FF; i += blockDim.x) {
    wk16[i] = (_Float16)kw[i];
    wq16[i] = (_Float16)qw[i];
  }
}

// -------- kernel 2: class-mean inputs, transposed to [C][T][F] f16 --------
__global__ void csa_classmean(const float* __restrict__ x, const int* __restrict__ labels,
                              const float* __restrict__ invcnt, _Float16* __restrict__ xct) {
  __shared__ int sLab[BB];
  int tid = threadIdx.x;
  int c = blockIdx.y;
  if (tid < BB) sLab[tid] = labels[tid];
  __syncthreads();
  int idx = blockIdx.x * blockDim.x + tid;   // [0, T*F)
  int t = idx & (TT - 1);
  int f = idx >> 9;                          // idx / TT
  float acc = 0.f;
  for (int b = 0; b < BB; ++b) {
    if (sLab[b] == c)                        // uniform branch: skips loads entirely
      acc += x[((size_t)(b * FF + f)) * TT + t];
  }
  xct[((size_t)(c * TT + t)) * FF + f] = (_Float16)(acc * invcnt[c]);
}

// -------- kernel 3: Q/K projection --------
// block = (c, which, t-slab). Async-stage the 16x128 A-slab in LDS once,
// 8 waves each compute one 16x16 d-tile: Out[t,d] = sum_f XcT[t,f]*W[d,f] + bias.
__global__ __launch_bounds__(256)
void csa_project(const _Float16* __restrict__ xct,
                 const _Float16* __restrict__ wq16, const _Float16* __restrict__ wk16,
                 const float* __restrict__ qb, const float* __restrict__ kb,
                 _Float16* __restrict__ qh, _Float16* __restrict__ kh) {
  __shared__ __align__(16) _Float16 sA[16 * FF];        // 4 KB
  int tid  = threadIdx.x;
  int lane = tid & 31;
  int wave = tid >> 5;
  int t0    = blockIdx.x * 16;
  int which = blockIdx.y;                                // 0 = Q, 1 = K
  int c     = blockIdx.z;
  int d0    = wave * 16;

  // stage A slab: 2048 halfs = 256 x 16B chunks, one per thread
  unsigned sA_base = lds_off_of(sA);
  async_copy_b128(sA_base + (unsigned)tid * 16u,
                  xct + ((size_t)c * TT + t0) * FF + tid * 8);
  wait_async0();
  __syncthreads();

  const _Float16* W = which ? wk16 : wq16;
  const float* bias = which ? kb : qb;
  _Float16* O = (which ? kh : qh) + (size_t)c * TT * DD;

  v8f acc = {};
#pragma unroll
  for (int kk = 0; kk < 4; ++kk) {
    half16 a = frag_a(sA + kk * 32, FF, lane);                      // ds_load_b128 x2
    half16 b = frag_b(W + (size_t)d0 * FF + kk * 32, FF, lane);     // global
    acc = WMMA_F16(a, b, acc);
  }
  int n = lane & 15, h = lane >> 4;
  float bv = bias[d0 + n];
#pragma unroll
  for (int r = 0; r < 8; ++r)
    O[(size_t)(t0 + r + 8 * h) * DD + d0 + n] = (_Float16)(acc[r] + bv);
}

// -------- kernel 4: fused A-tensor + abs-mean + softmax + row-mean --------
// block = (class i, 16-row t-block); async-stage the Q slab for ALL classes
// (10 x 16 x 128 f16 = 40 KB) once, then 8 waves sweep s-tiles; A[i,j] tiles
// never leave registers; attn slab + softmax in LDS.
__global__ __launch_bounds__(256)
void csa_attn(const _Float16* __restrict__ qh, const _Float16* __restrict__ kh,
              float* __restrict__ sm) {
  __shared__ __align__(16) _Float16 sQ[CC * 16 * DD];   // 40 KB
  __shared__ __align__(16) float sAttn[16 * TT];        // 32 KB
  int i = blockIdx.y;
  int t0 = blockIdx.x * 16;
  int tid = threadIdx.x;
  int wave = tid >> 5;
  int lane = tid & 31;

  // stage Q[t-slab] for all 10 classes: 10 x (256 x 16B chunks)
  unsigned sQ_base = lds_off_of(sQ);
#pragma unroll
  for (int j = 0; j < CC; ++j) {
    async_copy_b128(sQ_base + (unsigned)(j * 4096 + tid * 16),
                    qh + ((size_t)j * TT + t0) * DD + tid * 8);
  }
  wait_async0();
  __syncthreads();

  const _Float16* KhI = kh + (size_t)i * TT * DD;
  const _Float16* sQi = sQ + i * 16 * DD;

  for (int st = wave; st < TT / 16; st += 8) {
    int s0 = st * 16;
    // cache the 4 K-fragments of this s-tile: reused across all 10 j GEMMs
    half16 bf0 = frag_b(KhI + (size_t)s0 * DD + 0,  DD, lane);
    half16 bf1 = frag_b(KhI + (size_t)s0 * DD + 32, DD, lane);
    half16 bf2 = frag_b(KhI + (size_t)s0 * DD + 64, DD, lane);
    half16 bf3 = frag_b(KhI + (size_t)s0 * DD + 96, DD, lane);
    // diag tile: j == i
    v8f diag = {};
    diag = WMMA_F16(frag_a(sQi + 0,  DD, lane), bf0, diag);
    diag = WMMA_F16(frag_a(sQi + 32, DD, lane), bf1, diag);
    diag = WMMA_F16(frag_a(sQi + 64, DD, lane), bf2, diag);
    diag = WMMA_F16(frag_a(sQi + 96, DD, lane), bf3, diag);
    // mean_j |diag - A_j|  (j==i term is exactly 0)
    v8f asum = {};
    for (int j = 0; j < CC; ++j) {
      if (j == i) continue;                  // uniform branch, EXEC stays full for WMMA
      const _Float16* sQj = sQ + j * 16 * DD;
      v8f a = {};
      a = WMMA_F16(frag_a(sQj + 0,  DD, lane), bf0, a);
      a = WMMA_F16(frag_a(sQj + 32, DD, lane), bf1, a);
      a = WMMA_F16(frag_a(sQj + 64, DD, lane), bf2, a);
      a = WMMA_F16(frag_a(sQj + 96, DD, lane), bf3, a);
#pragma unroll
      for (int r = 0; r < 8; ++r) asum[r] += fabsf(diag[r] - a[r]);
    }
    int n = lane & 15, h = lane >> 4;
#pragma unroll
    for (int r = 0; r < 8; ++r)
      sAttn[(r + 8 * h) * TT + s0 + n] = diag[r] + asum[r] * (1.0f / CC);
  }
  __syncthreads();

  // softmax over s (512) per row t, then mean over s -> Sm[i, t0+row]
  for (int row = wave; row < 16; row += 8) {
    const float* rp = sAttn + row * TT;
    float v[16];
#pragma unroll
    for (int k = 0; k < 16; ++k) v[k] = rp[lane + 32 * k];
    float m = v[0];
#pragma unroll
    for (int k = 1; k < 16; ++k) m = fmaxf(m, v[k]);
    for (int off = 16; off; off >>= 1) m = fmaxf(m, __shfl_xor(m, off, 32));
    float z = 0.f;
#pragma unroll
    for (int k = 0; k < 16; ++k) z += __expf(v[k] - m);
    for (int off = 16; off; off >>= 1) z += __shfl_xor(z, off, 32);
    float invz = 1.0f / z;
    float ps = 0.f;
#pragma unroll
    for (int k = 0; k < 16; ++k) ps += __expf(v[k] - m) * invz;
    for (int off = 16; off; off >>= 1) ps += __shfl_xor(ps, off, 32);
    if (lane == 0) sm[i * TT + t0 + row] = ps * (1.0f / TT);
  }
}

// -------- kernel 5: out[b,c,f] = mean_t x[b,f,t] + sigma * sum_t x[b,f,t]*Sm[c,t] --------
__global__ __launch_bounds__(256)
void csa_out(const float* __restrict__ x, const float* __restrict__ sm,
             const float* __restrict__ sigma, float* __restrict__ out) {
  __shared__ float sSm[CC * TT];                 // 20 KB
  int tid = threadIdx.x;
  for (int k = tid; k < CC * TT; k += 256) sSm[k] = sm[k];
  __syncthreads();
  int lane = tid & 31;
  int g = blockIdx.x * 8 + (tid >> 5);           // one warp per (b, f) row
  int b = g >> 7;
  int f = g & (FF - 1);
  const float* rp = x + ((size_t)(b * FF + f)) * TT;
  float base = 0.f;
  float acc[CC];
#pragma unroll
  for (int c = 0; c < CC; ++c) acc[c] = 0.f;
  for (int t = lane; t < TT; t += 32) {
    float xv = rp[t];
    base += xv;
#pragma unroll
    for (int c = 0; c < CC; ++c) acc[c] += xv * sSm[c * TT + t];
  }
  for (int off = 16; off; off >>= 1) {
    base += __shfl_xor(base, off, 32);
#pragma unroll
    for (int c = 0; c < CC; ++c) acc[c] += __shfl_xor(acc[c], off, 32);
  }
  if (lane == 0) {
    float s = sigma[0];
    float bm = base * (1.0f / TT);
#pragma unroll
    for (int c = 0; c < CC; ++c)
      out[((size_t)(b * CC + c)) * FF + f] = bm + s * acc[c];
  }
}

extern "C" void kernel_launch(void* const* d_in, const int* in_sizes, int n_in,
                              void* d_out, int out_size, void* d_ws, size_t ws_size,
                              hipStream_t stream) {
  (void)in_sizes; (void)n_in; (void)out_size; (void)ws_size;
  const float* x   = (const float*)d_in[0];
  const float* kw  = (const float*)d_in[1];
  const float* kb  = (const float*)d_in[2];
  const float* qw  = (const float*)d_in[3];
  const float* qb  = (const float*)d_in[4];
  const float* sig = (const float*)d_in[5];
  const int*   lab = (const int*)d_in[6];
  // d_in[7] = iter_epoch, unused (training path is unconditional in reference)

  char* ws = (char*)d_ws;
  float*    invcnt = (float*)(ws + OFF_INV);
  _Float16* wq16   = (_Float16*)(ws + OFF_WQ);
  _Float16* wk16   = (_Float16*)(ws + OFF_WK);
  _Float16* xct    = (_Float16*)(ws + OFF_XCT);
  _Float16* qh     = (_Float16*)(ws + OFF_QH);
  _Float16* kh     = (_Float16*)(ws + OFF_KH);
  float*    sm     = (float*)(ws + OFF_SM);
  float*    out    = (float*)d_out;

  hipLaunchKernelGGL(csa_prep, dim3(1), dim3(256), 0, stream, kw, qw, lab, invcnt, wk16, wq16);
  hipLaunchKernelGGL(csa_classmean, dim3((TT * FF) / 256, CC), dim3(256), 0, stream,
                     x, lab, invcnt, xct);
  hipLaunchKernelGGL(csa_project, dim3(TT / 16, 2, CC), dim3(256), 0, stream,
                     xct, wq16, wk16, qb, kb, qh, kh);
  hipLaunchKernelGGL(csa_attn, dim3(TT / 16, CC), dim3(256), 0, stream, qh, kh, sm);
  hipLaunchKernelGGL(csa_out, dim3((BB * FF) / 8), dim3(256), 0, stream, x, sm, sig, out);
}
